// PhaseCoherentAttention_90632399880439
// MI455X (gfx1250) — matrix operand order repaired and checked by reading, not compiled, not run on hardware
//
#include <hip/hip_runtime.h>
#include <math.h>
#include <stdint.h>

// ---------------- problem constants ----------------
#define D_MODEL 192
#define NH      64
#define HQ      8
#define HKV     4
#define DH      24          // D_MODEL / HQ
#define B_SZ    2
#define S_LEN   2048
#define KDIM    160         // 24 (qk) + 64 (cos) + 64 (sin) + 8 pad = 5 * 32
#define SCALE_F 0.2041241452319315f   // 1/sqrt(24)
#define WAVES   4           // waves per attention workgroup

typedef __attribute__((ext_vector_type(16))) __bf16 v16bf;
typedef __attribute__((ext_vector_type(8)))  __bf16 v8bf;
typedef __attribute__((ext_vector_type(8)))  float  v8f;

// ---------------- WMMA helper ----------------
__device__ __forceinline__ v8f wmma_bf16(v16bf a, v16bf b, v8f c) {
    // D = A(16x32 bf16) * B(32x16 bf16) + C(16x16 f32)
    return __builtin_amdgcn_wmma_f32_16x16x32_bf16(
        false, a, false, b, (short)0, c, false, false);
}

// gather 16 bf16 as two 16B chunks (both pointers 16B aligned)
__device__ __forceinline__ v16bf load16bf(const __bf16* p0, const __bf16* p1) {
    v8bf lo = *(const v8bf*)p0;
    v8bf hi = *(const v8bf*)p1;
    v16bf r;
#pragma unroll
    for (int i = 0; i < 8; ++i) { r[i] = lo[i]; r[i + 8] = hi[i]; }
    return r;
}

// CDNA5 async global->LDS copy (16B per lane, tracked on ASYNCcnt)
__device__ __forceinline__ void async_load_b128(unsigned lds_byte_addr,
                                                unsigned long long gaddr) {
    asm volatile("global_load_async_to_lds_b128 %0, %1, off"
                 :: "v"(lds_byte_addr), "v"(gaddr) : "memory");
}
__device__ __forceinline__ void wait_asynccnt0() {
    asm volatile("s_wait_asynccnt 0" ::: "memory");
}
__device__ __forceinline__ void wait_dscnt0() {
    asm volatile("s_wait_dscnt 0" ::: "memory");
}
__device__ __forceinline__ unsigned lds_addr_of(const void* p) {
    // flat address of a __shared__ object: low 32 bits are the LDS byte offset
    return (unsigned)(uintptr_t)p;
}

// ============================================================
// Kernel 1: projections + phase features -> Qext / Kext / Vt
//   Qext[b][h][s][160]  bf16 : [a*scale*q | beta*cos/8 | beta*sin/8 | 0]
//   Kext[b][hk][s][160] bf16 : [k         | cos/8      | sin/8      | 0]
//   Vt  [b][hk][32][S]  bf16 : V transposed, dims 24..31 zero
// ============================================================
__global__ __launch_bounds__(D_MODEL)
void proj_kernel(const float* __restrict__ x,
                 const float* __restrict__ Wq,
                 const float* __restrict__ Wkv,
                 const float* __restrict__ alpha_p,
                 const float* __restrict__ beta_p,
                 __bf16* __restrict__ Qext,
                 __bf16* __restrict__ Kext,
                 __bf16* __restrict__ Vt) {
    const int row = blockIdx.x;            // b*S + s
    const int b = row / S_LEN;
    const int s = row % S_LEN;
    const int t = threadIdx.x;             // 0..191

    __shared__ float xr[D_MODEL];
    xr[t] = x[(size_t)row * D_MODEL + t];
    __syncthreads();

    const float alpha = alpha_p[0];
    const float beta  = beta_p[0];

    float accq = 0.f, acckv = 0.f;
    const float* wq  = Wq  + (size_t)t * D_MODEL;
    const float* wkv = Wkv + (size_t)t * D_MODEL;
#pragma unroll 8
    for (int j = 0; j < D_MODEL; ++j) {
        accq  = fmaf(wq[j],  xr[j], accq);
        acckv = fmaf(wkv[j], xr[j], acckv);
    }

    {   // Q part (all 192 outputs)
        const int h = t / DH, d = t % DH;
        Qext[(((size_t)b * HQ + h) * S_LEN + s) * KDIM + d] =
            (__bf16)(accq * alpha * SCALE_F);
    }
    if (t < 96) {                          // K
        const int h = t / DH, d = t % DH;
        Kext[(((size_t)b * HKV + h) * S_LEN + s) * KDIM + d] = (__bf16)acckv;
    } else {                               // V (transposed)
        const int u = t - 96;
        const int h = u / DH, d = u % DH;
        Vt[(((size_t)b * HKV + h) * 32 + d) * S_LEN + s] = (__bf16)acckv;
    }

    // phase features: channels 128..191, 1/8 per side -> 1/64 in the product
    if (t < NH) {
        const float ph = xr[2 * NH + t];
        const float c  = cosf(ph) * 0.125f;
        const float si = sinf(ph) * 0.125f;
#pragma unroll
        for (int h = 0; h < HQ; ++h) {
            __bf16* qr = Qext + (((size_t)b * HQ + h) * S_LEN + s) * KDIM;
            qr[DH + t]      = (__bf16)(beta * c);
            qr[DH + NH + t] = (__bf16)(beta * si);
        }
#pragma unroll
        for (int h = 0; h < HKV; ++h) {
            __bf16* kr = Kext + (((size_t)b * HKV + h) * S_LEN + s) * KDIM;
            kr[DH + t]      = (__bf16)c;
            kr[DH + NH + t] = (__bf16)si;
        }
    }
    if (t >= 64 && t < 72) {               // zero pads dims 152..159
        const int d = 152 + (t - 64);
#pragma unroll
        for (int h = 0; h < HQ; ++h)
            Qext[(((size_t)b * HQ + h) * S_LEN + s) * KDIM + d] = (__bf16)0.f;
#pragma unroll
        for (int h = 0; h < HKV; ++h)
            Kext[(((size_t)b * HKV + h) * S_LEN + s) * KDIM + d] = (__bf16)0.f;
    }
    if (t >= 96 && t < 128) {              // zero pad Vt dims 24..31
        const int u = t - 96;
        const int h = u / 8, d = 24 + (u % 8);
        Vt[(((size_t)b * HKV + h) * 32 + d) * S_LEN + s] = (__bf16)0.f;
    }
}

// ============================================================
// Kernel 2: flash attention, 4 waves / WG sharing K/V via
// double-buffered async global->LDS copies.
//   wave w handles q-tile blockIdx.x*4 + w of head blockIdx.y
// ============================================================
__global__ __launch_bounds__(32 * WAVES)
void attn_kernel(const __bf16* __restrict__ Qext,
                 const __bf16* __restrict__ Kext,
                 const __bf16* __restrict__ Vt,
                 float* __restrict__ Oattn) {
    const int h  = blockIdx.y;
    const int b  = blockIdx.z;
    const int hk = h >> 1;                 // GQA: jnp.repeat(k,2,axis=2)

    const int tid   = threadIdx.x;         // 0..127
    // wave id is uniform across the wave: pin it into an SGPR so the causal
    // participation test becomes an s_cbranch (EXEC stays all-1s for WMMA).
    const int w     = __builtin_amdgcn_readfirstlane(tid >> 5);
    const int lane  = tid & 31;
    const int lmod  = lane & 15;
    const int lhalf = lane >> 4;
    const int q0    = (blockIdx.x * WAVES + w) * 16;   // uniform (SGPR)

    const __bf16* Qb = Qext + (((size_t)b * HQ  + h ) * S_LEN + q0) * KDIM;
    const __bf16* Kb = Kext + (((size_t)b * HKV + hk) * S_LEN) * KDIM;
    const __bf16* Vb = Vt   + (((size_t)b * HKV + hk) * 32) * S_LEN;

    // double-buffered K/V slabs + per-wave P staging
    __shared__ __align__(16) __bf16 Ksh[2][32 * KDIM];   // 2 x 10 KB
    __shared__ __align__(16) __bf16 Vsh[2][32 * 32];     // 2 x 2 KB
    __shared__ __align__(16) float  Pst[WAVES][16][32];  // 8 KB

    // Q tile as 5 A-fragments (16x32 bf16, contiguous k = 16*lhalf + s)
    v16bf qa[5];
#pragma unroll
    for (int c = 0; c < 5; ++c) {
        const __bf16* p = Qb + (size_t)lmod * KDIM + c * 32 + 16 * lhalf;
        qa[c] = load16bf(p, p + 8);
    }

    v8f o0 = {}; v8f o1 = {};
    float mrow[8], lrow[8];
#pragma unroll
    for (int r = 0; r < 8; ++r) { mrow[r] = -3.0e38f; lrow[r] = 0.f; }

    // causal extents: block iterates to the last wave's limit
    const int q0_last = (blockIdx.x * WAVES + (WAVES - 1)) * 16;
    const int nkb     = (q0_last + 15) / 32 + 1;

    // cooperative async issue of one 32-key block into buffer `bufsel`
    auto issue_block = [&](int kb, int bufsel) {
        // K slab: 32 keys x 160 bf16 = 10240 contiguous bytes
        const unsigned long long kg =
            (unsigned long long)(uintptr_t)Kb + (unsigned long long)kb * (KDIM * 2);
        const unsigned ks = lds_addr_of(&Ksh[bufsel][0]);
#pragma unroll
        for (int j = 0; j < 5; ++j) {
            const int off = (tid + 128 * j) * 16;
            async_load_b128(ks + off, kg + off);
        }
        // V slab: 32 dim-rows x 64 bytes, global row stride S*2
        const int vrow = tid >> 2, vchunk = (tid & 3) * 16;
        const unsigned long long vg =
            (unsigned long long)(uintptr_t)Vb +
            (unsigned long long)vrow * (S_LEN * 2) + (unsigned long long)kb * 2 + vchunk;
        async_load_b128(lds_addr_of(&Vsh[bufsel][0]) + vrow * 64 + vchunk, vg);
    };

    issue_block(0, 0);                      // preload first block

    for (int kb32 = 0; kb32 < nkb; ++kb32) {
        const int kb  = kb32 * 32;
        const int cur = kb32 & 1;

        wait_asynccnt0();                   // my async stores to LDS done
        __syncthreads();                    // everyone's done; prev reads done

        if (kb32 + 1 < nkb) issue_block(kb + 32, cur ^ 1);  // overlap next copy

        if (kb <= q0 + 15) {                // SGPR compare -> s_cbranch skip
            // ---- scores: two 16x16 tiles over 5 K-chunks (10 WMMAs) ----
            v8f c0 = {}; v8f c1 = {};
#pragma unroll
            for (int c = 0; c < 5; ++c) {
                const __bf16* pk0 = &Ksh[cur][(size_t)lmod * KDIM + c * 32 + lhalf * 16];
                const __bf16* pk1 = pk0 + (size_t)16 * KDIM;
                v16bf b0 = load16bf(pk0, pk0 + 8);
                v16bf b1 = load16bf(pk1, pk1 + 8);
                c0 = wmma_bf16(qa[c], b0, c0);
                c1 = wmma_bf16(qa[c], b1, c1);
            }

            // ---- causal mask (uniform test; per-lane cndmask inside) ----
            if (kb + 31 > q0) {
#pragma unroll
                for (int r = 0; r < 8; ++r) {
                    const int qi = q0 + r + 8 * lhalf;
                    if (kb + lmod      > qi) c0[r] = -1.0e9f;
                    if (kb + 16 + lmod > qi) c1[r] = -1.0e9f;
                }
            }

            // ---- online softmax (row reductions within 16-lane halves) ----
            float p0[8], p1[8];
#pragma unroll
            for (int r = 0; r < 8; ++r) {
                float rm = fmaxf(c0[r], c1[r]);
                rm = fmaxf(rm, __shfl_xor(rm, 1));
                rm = fmaxf(rm, __shfl_xor(rm, 2));
                rm = fmaxf(rm, __shfl_xor(rm, 4));
                rm = fmaxf(rm, __shfl_xor(rm, 8));
                const float mnew  = fmaxf(mrow[r], rm);
                const float scale = __expf(mrow[r] - mnew);
                p0[r] = __expf(c0[r] - mnew);
                p1[r] = __expf(c1[r] - mnew);
                float rs = p0[r] + p1[r];
                rs += __shfl_xor(rs, 1);
                rs += __shfl_xor(rs, 2);
                rs += __shfl_xor(rs, 4);
                rs += __shfl_xor(rs, 8);
                lrow[r] = lrow[r] * scale + rs;
                mrow[r] = mnew;
                o0[r] *= scale; o1[r] *= scale;
            }

            // ---- P (C layout) -> A-fragment via per-wave LDS staging ----
#pragma unroll
            for (int r = 0; r < 8; ++r) {
                Pst[w][r + 8 * lhalf][lmod]      = p0[r];
                Pst[w][r + 8 * lhalf][lmod + 16] = p1[r];
            }
            wait_dscnt0();                  // wave-local: DS ops are in-order
            v16bf pa;
#pragma unroll
            for (int j = 0; j < 8; ++j) {
                pa[j]     = (__bf16)Pst[w][lmod][16 * lhalf + j];
                pa[j + 8] = (__bf16)Pst[w][lmod][16 * lhalf + 8 + j];
            }

            // ---- AV: P(16x32) x V(32x16) for dim chunks 0..15 / 16..31 ----
            const __bf16* pv0 = &Vsh[cur][(size_t)(0  + lmod) * 32 + lhalf * 16];
            const __bf16* pv1 = &Vsh[cur][(size_t)(16 + lmod) * 32 + lhalf * 16];
            v16bf vb0 = load16bf(pv0, pv0 + 8);
            v16bf vb1 = load16bf(pv1, pv1 + 8);
            o0 = wmma_bf16(pa, vb0, o0);
            o1 = wmma_bf16(pa, vb1, o1);
        }
    }

    // ---- normalize and write (only dims 0..23 are real) ----
#pragma unroll
    for (int r = 0; r < 8; ++r) {
        const int m = r + 8 * lhalf;
        const float inv = 1.0f / lrow[r];
        float* orow = Oattn + (((size_t)b * S_LEN) + q0 + m) * D_MODEL + h * DH;
        orow[lmod] = o0[r] * inv;
        if (lmod < 8) orow[16 + lmod] = o1[r] * inv;
    }
}

// ============================================================
// Kernel 3: output projection  out = Oattn @ Wo^T
// ============================================================
__global__ __launch_bounds__(D_MODEL)
void out_kernel(const float* __restrict__ Oattn,
                const float* __restrict__ Wo,
                float* __restrict__ out) {
    const int row = blockIdx.x;
    const int t   = threadIdx.x;
    __shared__ float orow[D_MODEL];
    orow[t] = Oattn[(size_t)row * D_MODEL + t];
    __syncthreads();
    float acc = 0.f;
    const float* wp = Wo + (size_t)t * D_MODEL;
#pragma unroll 8
    for (int j = 0; j < D_MODEL; ++j) acc = fmaf(wp[j], orow[j], acc);
    out[(size_t)row * D_MODEL + t] = acc;
}

// ============================================================
extern "C" void kernel_launch(void* const* d_in, const int* in_sizes, int n_in,
                              void* d_out, int out_size, void* d_ws, size_t ws_size,
                              hipStream_t stream) {
    const float* x     = (const float*)d_in[0];
    const float* Wq    = (const float*)d_in[1];
    const float* Wkv   = (const float*)d_in[2];
    const float* Wo    = (const float*)d_in[3];
    const float* alpha = (const float*)d_in[4];
    const float* beta  = (const float*)d_in[5];
    float* out = (float*)d_out;

    char* ws = (char*)d_ws;
    size_t off = 0;
    auto carve = [&](size_t bytes) -> void* {
        void* p = ws + off;
        off = (off + bytes + 255) & ~(size_t)255;
        return p;
    };
    const size_t QEXT_B = (size_t)B_SZ * HQ  * S_LEN * KDIM * sizeof(__bf16);
    const size_t KEXT_B = (size_t)B_SZ * HKV * S_LEN * KDIM * sizeof(__bf16);
    const size_t VT_B   = (size_t)B_SZ * HKV * 32    * S_LEN * sizeof(__bf16);
    const size_t OAT_B  = (size_t)B_SZ * S_LEN * D_MODEL * sizeof(float);

    __bf16* Qext = (__bf16*)carve(QEXT_B);
    __bf16* Kext = (__bf16*)carve(KEXT_B);
    __bf16* Vt   = (__bf16*)carve(VT_B);
    float*  Oatt = (float*)carve(OAT_B);

    proj_kernel<<<dim3(B_SZ * S_LEN), dim3(D_MODEL), 0, stream>>>(
        x, Wq, Wkv, alpha, beta, Qext, Kext, Vt);

    attn_kernel<<<dim3(S_LEN / (16 * WAVES), HQ, B_SZ), dim3(32 * WAVES), 0, stream>>>(
        Qext, Kext, Vt, Oatt);

    out_kernel<<<dim3(B_SZ * S_LEN), dim3(D_MODEL), 0, stream>>>(Oatt, Wo, out);
}
